// SpikingSynapticResonance_25262997636011
// MI455X (gfx1250) — compile-verified, matrix operand order checked by reading, not run on hardware
//
#include <hip/hip_runtime.h>
#include <hip/hip_bf16.h>
#include <math.h>

typedef __attribute__((ext_vector_type(16))) _Float16 v16h;
typedef __attribute__((ext_vector_type(8)))  float    v8f;
typedef __attribute__((ext_vector_type(2)))  _Float16 h2;

#define T_TOTAL 4
#define BATCH   2
#define SEQ     2048
#define DMODEL  1024
#define NHEADS  16
#define DHEAD   64
#define TOPK    64

__device__ __forceinline__ int lane_id() { return threadIdx.x & 31; }

__device__ __forceinline__ v8f zero8() {
  v8f z;
#pragma unroll
  for (int i = 0; i < 8; ++i) z[i] = 0.0f;
  return z;
}

// First K index of the halfword pair held in fragment register-half j.
// Layout per CDNA5 ISA 7.12.2 (16-bit A-matrix 16x32; B uses the mirrored
// N-major / K-contiguous gather).
__device__ __forceinline__ int frag_k0(int j, int grp) {
  return ((j >> 2) << 4) + (grp << 3) + ((j & 3) << 1);
}

// Load a 16x32 f16 fragment from row-major f16 memory (K contiguous).
// Per lane this is two contiguous 16B runs -> two b128 loads after merging.
__device__ __forceinline__ v16h load_frag_f16(const _Float16* __restrict__ base, int ld) {
  const int l = lane_id(), row = l & 15, grp = l >> 4;
  const _Float16* p = base + (size_t)row * ld;
  v16h a;
#pragma unroll
  for (int j = 0; j < 8; ++j) {
    const int k0 = frag_k0(j, grp);
    const h2 f = *(const h2*)(p + k0);
    a[2 * j]     = f.x;
    a[2 * j + 1] = f.y;
  }
  return a;
}

__device__ __forceinline__ v8f wmma16(v16h a, v16h b, v8f c) {
  return __builtin_amdgcn_wmma_f32_16x16x32_f16(false, a, false, b, (short)0, c,
                                                false, false);
}

// ---------------------------------------------------------------------------
// Kernel 0: bulk f32 -> f16 conversion (bandwidth-bound, one pass).
// ---------------------------------------------------------------------------
__global__ void __launch_bounds__(256)
f32_to_f16_kernel(const float* __restrict__ src, _Float16* __restrict__ dst, int n4) {
  const int i = blockIdx.x * blockDim.x + threadIdx.x;
  if (i >= n4) return;
  const float4 f = ((const float4*)src)[i];
  h2 a, b;
  a.x = (_Float16)f.x; a.y = (_Float16)f.y;
  b.x = (_Float16)f.z; b.y = (_Float16)f.w;
  ((h2*)dst)[2 * i]     = a;
  ((h2*)dst)[2 * i + 1] = b;
}

// ---------------------------------------------------------------------------
// Kernel 1: fused QKV GEMM (f16 WMMA) + LIF over T + temporal softmax mix.
// 4 waves / WG, each wave owns one 16x16 (rows x e-cols) tile across all T.
// ---------------------------------------------------------------------------
__global__ void __launch_bounds__(128)
qkv_lif_mix_kernel(const _Float16* __restrict__ xh,  // (T,B,S,D) f16 spikes
                   const _Float16* __restrict__ Wqh,
                   const _Float16* __restrict__ Wkh,
                   const _Float16* __restrict__ Wvh,
                   const float* __restrict__ beta_q,
                   const float* __restrict__ beta_k,
                   const float* __restrict__ tmix_q,
                   const float* __restrict__ tmix_k,
                   float* __restrict__ qmix,         // (B,H,S,Dh) f32
                   float* __restrict__ kmix,         // (B,H,S,Dh) f32
                   _Float16* __restrict__ vT)        // (B,H,Dh,S) f16
{
  const int wave    = threadIdx.x >> 5;
  const int nbase   = blockIdx.x * 64 + wave * 16;
  const int rowbase = blockIdx.y * 16;              // rows of flattened (b,s)

  v8f accq[T_TOTAL], acck[T_TOTAL], accv[T_TOTAL];
#pragma unroll
  for (int t = 0; t < T_TOTAL; ++t) { accq[t] = zero8(); acck[t] = zero8(); accv[t] = zero8(); }

  const size_t xrow = (size_t)rowbase * DMODEL;
  for (int kb = 0; kb < DMODEL; kb += 32) {
    const v16h bq = load_frag_f16(Wqh + (size_t)nbase * DMODEL + kb, DMODEL);
    const v16h bk = load_frag_f16(Wkh + (size_t)nbase * DMODEL + kb, DMODEL);
    const v16h bv = load_frag_f16(Wvh + (size_t)nbase * DMODEL + kb, DMODEL);
#pragma unroll
    for (int t = 0; t < T_TOTAL; ++t) {
      const _Float16* xa = xh + (size_t)t * BATCH * SEQ * DMODEL + xrow + kb;
      __builtin_prefetch(xa + 32, 0, 1);            // next K-tile -> global_prefetch
      const v16h a = load_frag_f16(xa, DMODEL);
      accq[t] = wmma16(a, bq, accq[t]);
      acck[t] = wmma16(a, bk, acck[t]);
      accv[t] = wmma16(a, bv, accv[t]);
    }
  }

  const int l = lane_id(), grp = l >> 4, col = l & 15;
  const int e = nbase + col;
  const float bq_e = beta_q[e], bk_e = beta_k[e];

  // softmax of temporal mixing weights (4 scalars)
  float wq[T_TOTAL], wk[T_TOTAL];
  {
    float mq = tmix_q[0], mk = tmix_k[0];
#pragma unroll
    for (int t = 1; t < T_TOTAL; ++t) { mq = fmaxf(mq, tmix_q[t]); mk = fmaxf(mk, tmix_k[t]); }
    float sq = 0.f, sk = 0.f;
#pragma unroll
    for (int t = 0; t < T_TOTAL; ++t) {
      wq[t] = __expf(tmix_q[t] - mq); sq += wq[t];
      wk[t] = __expf(tmix_k[t] - mk); sk += wk[t];
    }
#pragma unroll
    for (int t = 0; t < T_TOTAL; ++t) { wq[t] /= sq; wk[t] /= sk; }
  }

  const int h = e >> 6, dh = e & 63;
#pragma unroll
  for (int j = 0; j < 8; ++j) {
    const int m = j + (grp << 3);
    const int r = rowbase + m;
    const int b = r / SEQ, s = r % SEQ;
    float vq = 0.f, vk = 0.f, qm = 0.f, km = 0.f, vm = 0.f;
#pragma unroll
    for (int t = 0; t < T_TOTAL; ++t) {
      vq = bq_e * vq + accq[t][j];                   // LIF membrane
      const float sq_ = (vq >= 1.0f) ? 1.0f : 0.0f;  // forward = hard spike
      vq -= sq_;
      qm += wq[t] * sq_;
      vk = bk_e * vk + acck[t][j];
      const float sk_ = (vk >= 1.0f) ? 1.0f : 0.0f;
      vk -= sk_;
      km += wk[t] * sk_;
      vm += accv[t][j];
    }
    vm *= (1.0f / T_TOTAL);
    const size_t qi = (((size_t)b * NHEADS + h) * SEQ + s) * DHEAD + dh;
    qmix[qi] = qm;
    kmix[qi] = km;
    vT[(((size_t)b * NHEADS + h) * DHEAD + dh) * SEQ + s] = (_Float16)vm;
  }
}

// ---------------------------------------------------------------------------
// Kernel 2: RoPE on mixed q/k, emit f16 tensors for the attention WMMAs.
// ---------------------------------------------------------------------------
__global__ void __launch_bounds__(256)
rope_kernel(const float* __restrict__ qmix, const float* __restrict__ kmix,
            _Float16* __restrict__ qh, _Float16* __restrict__ kh)
{
  const int idx = blockIdx.x * blockDim.x + threadIdx.x;    // B*H*S*(Dh/2)
  if (idx >= BATCH * NHEADS * SEQ * (DHEAD / 2)) return;
  const int i  = idx & 31;
  const int s  = (idx >> 5) & (SEQ - 1);
  const int bh = idx >> 16;
  const size_t base = ((size_t)bh * SEQ + s) * DHEAD;
  // inv_freq = theta^(-2i/64), log(10000) = 9.2103403...
  const float inv_f = __expf(-(float)(2 * i) * (9.210340371976184f / 64.0f));
  const float ang = (float)s * inv_f;
  float sn, c;
  __sincosf(ang, &sn, &c);
  float x1 = qmix[base + i], x2 = qmix[base + i + 32];
  qh[base + i]      = (_Float16)(x1 * c - x2 * sn);
  qh[base + i + 32] = (_Float16)(x1 * sn + x2 * c);
  x1 = kmix[base + i]; x2 = kmix[base + i + 32];
  kh[base + i]      = (_Float16)(x1 * c - x2 * sn);
  kh[base + i + 32] = (_Float16)(x1 * sn + x2 * c);
}

// ---------------------------------------------------------------------------
// Kernel 3: causal attention with top-64 sparsification.
// 4 waves per WG handle 16 query rows vs all 2048 keys; 128 KB dynamic LDS
// holds the 16 x 2048 f32 score matrix (gfx1250: 320 KB per WG). Softmax
// probs are written back to LDS as f16 so PV A-fragments need no cvt.
// ---------------------------------------------------------------------------
__global__ void __launch_bounds__(128)
attention_kernel(const _Float16* __restrict__ qh,  // (B,H,S,Dh)
                 const _Float16* __restrict__ kh,  // (B,H,S,Dh)
                 const _Float16* __restrict__ vT,  // (B,H,Dh,S)
                 const float* __restrict__ temp_p,
                 _Float16* __restrict__ ctx)       // (B,S,D) f16
{
  extern __shared__ float sc[];                    // [16][SEQ]
  const int wave = threadIdx.x >> 5;
  const int l = lane_id(), grp = l >> 4, col = l & 15;
  const int bh = blockIdx.y;
  const int b = bh >> 4, h = bh & 15;
  const int qbase = blockIdx.x * 16;
  const float temp = temp_p[0];
  const float NEGINF = -__builtin_inff();

  const _Float16* qp = qh + ((size_t)bh * SEQ + qbase) * DHEAD;
  const v16h aq0 = load_frag_f16(qp, DHEAD);       // Dh 0..31
  const v16h aq1 = load_frag_f16(qp + 32, DHEAD);  // Dh 32..63

  // ---- phase 1: scores = (q . k^T) * temp, causal mask -> LDS (f32) ----
  for (int kt = wave * 32; kt < wave * 32 + 32; ++kt) {
    const int kbase = kt * 16;
    const _Float16* kp = kh + ((size_t)bh * SEQ + kbase) * DHEAD;
    const v16h b0 = load_frag_f16(kp, DHEAD);
    const v16h b1 = load_frag_f16(kp + 32, DHEAD);
    v8f acc = zero8();
    acc = wmma16(aq0, b0, acc);
    acc = wmma16(aq1, b1, acc);
    const int ks = kbase + col;
#pragma unroll
    for (int j = 0; j < 8; ++j) {
      const int m = j + (grp << 3);
      const int qs = qbase + m;
      sc[m * SEQ + ks] = (ks <= qs) ? acc[j] * temp : NEGINF;
    }
  }
  __syncthreads();

  // ---- phase 2: per-row top-64 (radix select) + softmax; write f16 probs ----
  for (int m = wave; m < 16; m += 4) {
    float* row = sc + m * SEQ;
    const int qs = qbase + m;
    const int valid = qs + 1;                      // causal-valid entries
    float v[64];
#pragma unroll
    for (int j = 0; j < 64; ++j) v[j] = row[j * 32 + l];
    float mx = NEGINF;
#pragma unroll
    for (int j = 0; j < 64; ++j) mx = fmaxf(mx, v[j]);
    for (int off = 16; off > 0; off >>= 1) mx = fmaxf(mx, __shfl_xor(mx, off, 32));

    unsigned thr = 0u;
    if (valid > TOPK) {
      unsigned uv[64];
#pragma unroll
      for (int j = 0; j < 64; ++j) {               // monotone float->uint map
        const unsigned bits = __float_as_uint(v[j]);
        uv[j] = (bits & 0x80000000u) ? ~bits : (bits | 0x80000000u);
      }
      for (int bit = 31; bit >= 0; --bit) {        // binary search 64th largest
        const unsigned cand = thr | (1u << bit);
        int cnt = 0;
#pragma unroll
        for (int j = 0; j < 64; ++j) cnt += (uv[j] >= cand) ? 1 : 0;
        for (int off = 16; off > 0; off >>= 1) cnt += __shfl_xor(cnt, off, 32);
        if (cnt >= TOPK) thr = cand;
      }
    }
    float p[64];
    float sum = 0.f;
#pragma unroll
    for (int j = 0; j < 64; ++j) {
      bool keep;
      if (valid > TOPK) {
        const unsigned bits = __float_as_uint(v[j]);
        const unsigned u = (bits & 0x80000000u) ? ~bits : (bits | 0x80000000u);
        keep = (u >= thr);
      } else {
        keep = (v[j] != NEGINF);
      }
      p[j] = keep ? __expf(v[j] - mx) : 0.f;
      sum += p[j];
    }
    for (int off = 16; off > 0; off >>= 1) sum += __shfl_xor(sum, off, 32);
    const float inv = 1.0f / sum;
    // write probs back as f16, packed into the first half of this row
    _Float16* rowh = (_Float16*)row;
#pragma unroll
    for (int j = 0; j < 64; ++j) rowh[j * 32 + l] = (_Float16)(p[j] * inv);
  }
  __syncthreads();

  // ---- phase 3: ctx = probs @ V, each wave owns one Dh 16-col tile ----
  const int nb = wave * 16;
  v8f acc = zero8();
  const _Float16* vp  = vT + ((size_t)bh * DHEAD + nb) * SEQ;
  const _Float16* sch = (const _Float16*)sc;       // f16 probs, row stride 2*SEQ
  for (int kk = 0; kk < SEQ; kk += 32) {
    const v16h a   = load_frag_f16(sch + kk, 2 * SEQ);  // probs from LDS
    const v16h bfr = load_frag_f16(vp + kk, SEQ);       // V^T, K-pairs contiguous
    acc = wmma16(a, bfr, acc);
  }
#pragma unroll
  for (int j = 0; j < 8; ++j) {
    const int m = j + (grp << 3);
    const int s = qbase + m;
    const int e = h * DHEAD + nb + col;
    ctx[((size_t)b * SEQ + s) * DMODEL + e] = (_Float16)acc[j];
  }
}

// ---------------------------------------------------------------------------
// Kernel 4: out = ctx @ W_o^T, broadcast over T.
// ---------------------------------------------------------------------------
__global__ void __launch_bounds__(128)
outproj_kernel(const _Float16* __restrict__ ctx,   // (B,S,D) f16
               const _Float16* __restrict__ Woh,   // (D,D) f16
               float* __restrict__ out)            // (T,B,S,D) f32
{
  const int wave    = threadIdx.x >> 5;
  const int nbase   = blockIdx.x * 64 + wave * 16;
  const int rowbase = blockIdx.y * 16;
  const int l = lane_id(), grp = l >> 4, col = l & 15;
  v8f acc = zero8();
  for (int kb = 0; kb < DMODEL; kb += 32) {
    const v16h a   = load_frag_f16(ctx + (size_t)rowbase * DMODEL + kb, DMODEL);
    const v16h bfr = load_frag_f16(Woh + (size_t)nbase * DMODEL + kb, DMODEL);
    acc = wmma16(a, bfr, acc);
  }
#pragma unroll
  for (int j = 0; j < 8; ++j) {
    const int r = rowbase + j + (grp << 3);
    const size_t o = (size_t)r * DMODEL + nbase + col;
    const float val = acc[j];
#pragma unroll
    for (int t = 0; t < T_TOTAL; ++t)
      out[(size_t)t * BATCH * SEQ * DMODEL + o] = val;
  }
}

extern "C" void kernel_launch(void* const* d_in, const int* in_sizes, int n_in,
                              void* d_out, int out_size, void* d_ws, size_t ws_size,
                              hipStream_t stream) {
  const float* x    = (const float*)d_in[0];
  const float* Wq   = (const float*)d_in[1];
  const float* Wk   = (const float*)d_in[2];
  const float* Wv   = (const float*)d_in[3];
  const float* Wo   = (const float*)d_in[4];
  const float* bq   = (const float*)d_in[5];
  const float* bk   = (const float*)d_in[6];
  const float* temp = (const float*)d_in[7];
  const float* tq   = (const float*)d_in[8];
  const float* tk   = (const float*)d_in[9];
  float* out = (float*)d_out;

  const size_t BSD  = (size_t)BATCH * SEQ * DMODEL;   // 4M elements
  const size_t TBSD = (size_t)T_TOTAL * BSD;          // 16.7M elements
  const size_t DD   = (size_t)DMODEL * DMODEL;        // 1M elements

  char* w = (char*)d_ws;
  _Float16* xhb  = (_Float16*)w;  w += TBSD * sizeof(_Float16);
  _Float16* Wqh  = (_Float16*)w;  w += DD * sizeof(_Float16);
  _Float16* Wkh  = (_Float16*)w;  w += DD * sizeof(_Float16);
  _Float16* Wvh  = (_Float16*)w;  w += DD * sizeof(_Float16);
  _Float16* Woh  = (_Float16*)w;  w += DD * sizeof(_Float16);
  float*    qmix = (float*)w;     w += BSD * sizeof(float);
  float*    kmix = (float*)w;     w += BSD * sizeof(float);
  _Float16* qhh  = (_Float16*)w;  w += BSD * sizeof(_Float16);
  _Float16* khh  = (_Float16*)w;  w += BSD * sizeof(_Float16);
  _Float16* vT   = (_Float16*)w;  w += BSD * sizeof(_Float16);
  _Float16* ctx  = (_Float16*)w;  w += BSD * sizeof(_Float16);

  // prep: one-pass f32->f16 of activations and all weights
  f32_to_f16_kernel<<<(int)((TBSD / 4 + 255) / 256), 256, 0, stream>>>(x, xhb, (int)(TBSD / 4));
  f32_to_f16_kernel<<<(int)((DD / 4 + 255) / 256), 256, 0, stream>>>(Wq, Wqh, (int)(DD / 4));
  f32_to_f16_kernel<<<(int)((DD / 4 + 255) / 256), 256, 0, stream>>>(Wk, Wkh, (int)(DD / 4));
  f32_to_f16_kernel<<<(int)((DD / 4 + 255) / 256), 256, 0, stream>>>(Wv, Wvh, (int)(DD / 4));
  f32_to_f16_kernel<<<(int)((DD / 4 + 255) / 256), 256, 0, stream>>>(Wo, Woh, (int)(DD / 4));

  dim3 g1(DMODEL / 64, (BATCH * SEQ) / 16);
  qkv_lif_mix_kernel<<<g1, 128, 0, stream>>>(xhb, Wqh, Wkh, Wvh, bq, bk, tq, tk,
                                             qmix, kmix, vT);

  const int nrope = BATCH * NHEADS * SEQ * (DHEAD / 2);
  rope_kernel<<<(nrope + 255) / 256, 256, 0, stream>>>(qmix, kmix, qhh, khh);

  dim3 g3(SEQ / 16, BATCH * NHEADS);
  attention_kernel<<<g3, 128, 16 * SEQ * sizeof(float), stream>>>(qhh, khh, vT,
                                                                  temp, ctx);

  dim3 g4(DMODEL / 64, (BATCH * SEQ) / 16);
  outproj_kernel<<<g4, 128, 0, stream>>>(ctx, Woh, out);
}
